// AttentionRCNN_22007412425199
// MI455X (gfx1250) — compile-verified, matrix operand order checked by reading, not run on hardware
//
#include <hip/hip_runtime.h>
#include <hip/hip_bf16.h>

// Problem constants (from reference)
constexpr int Bb = 4, Nn = 4, Cc = 256, Hh = 100, Ww = 152;
constexpr int HW  = Hh * Ww;        // 15200
constexpr int PIX = Bb * HW;        // 60800  (divisible by 64)

typedef __attribute__((ext_vector_type(16))) __bf16 v16bf;
typedef __attribute__((ext_vector_type(8)))  float  v8f;

// ---------------------------------------------------------------------------
// 1) Per-pixel channel sums: cs[m][b][p], m=0 -> features, m=1..4 -> nearby n
// ---------------------------------------------------------------------------
__global__ __launch_bounds__(256) void k_chansum(const float* __restrict__ f,
                                                 const float* __restrict__ nb,
                                                 float* __restrict__ cs) {
    int idx = blockIdx.x * 256 + threadIdx.x;          // over 5*B*HW
    if (idx >= 5 * Bb * HW) return;
    int m = idx / (Bb * HW);
    int r = idx - m * (Bb * HW);
    int b = r / HW;
    int p = r - b * HW;
    const float* src = (m == 0)
        ? (f  + ((size_t)b * Cc) * HW + p)
        : (nb + (((size_t)b * Nn + (m - 1)) * Cc) * HW + p);
    float s = 0.f;
    for (int c = 0; c < Cc; ++c) s += src[(size_t)c * HW];
    cs[idx] = s;
}

// ---------------------------------------------------------------------------
// 2) 3x3 edge-clamped box average: avg[m][b][p] = boxsum(cs)/(9*C)
// ---------------------------------------------------------------------------
__global__ __launch_bounds__(256) void k_avg(const float* __restrict__ cs,
                                             float* __restrict__ avg) {
    int idx = blockIdx.x * 256 + threadIdx.x;          // over 5*B*HW
    if (idx >= 5 * Bb * HW) return;
    int mb = idx / HW;
    int p  = idx - mb * HW;
    int i  = p / Ww, j = p - i * Ww;
    const float* s = cs + (size_t)mb * HW;
    float sum = 0.f;
#pragma unroll
    for (int di = -1; di <= 1; ++di) {
        int gi = min(max(i + di, 0), Hh - 1);
#pragma unroll
        for (int dj = -1; dj <= 1; ++dj) {
            int gj = min(max(j + dj, 0), Ww - 1);
            sum += s[gi * Ww + gj];
        }
    }
    avg[idx] = sum * (1.0f / (9.0f * Cc));
}

// ---------------------------------------------------------------------------
// 3) Census similarity. Tile 8x32 pixels per block; LDS halo per channel.
//    Feature-side comparison (t1) computed once, shared across all N=4.
// ---------------------------------------------------------------------------
constexpr int TI = 8, TJ = 32;
constexpr int HALO = (TI + 2) * (TJ + 2);              // 340

__global__ __launch_bounds__(256) void k_sim(const float* __restrict__ f,
                                             const float* __restrict__ nb,
                                             const float* __restrict__ avg,
                                             float* __restrict__ sim) {
    __shared__ float lf[HALO];
    __shared__ float ln[Nn][HALO];
    const int b  = blockIdx.z;
    const int i0 = blockIdx.y * TI;
    const int j0 = blockIdx.x * TJ;
    const int ti = threadIdx.x / TJ;
    const int tj = threadIdx.x - ti * TJ;
    const int i  = i0 + ti, j = j0 + tj;
    const bool valid = (i < Hh) && (j < Ww);
    const int  p = i * Ww + j;

    float a1 = 0.f, a2[Nn] = {0.f, 0.f, 0.f, 0.f};
    if (valid) {
        a1 = avg[(size_t)b * HW + p];                       // m = 0
#pragma unroll
        for (int n = 0; n < Nn; ++n)
            a2[n] = avg[((size_t)(1 + n) * Bb + b) * HW + p];
    }
    int cnt[Nn] = {0, 0, 0, 0};
    const size_t fb = (size_t)b * Cc * HW;
    const size_t nbb = (size_t)b * Nn * Cc * HW;

    for (int c = 0; c < Cc; ++c) {
        for (int idx = threadIdx.x; idx < HALO; idx += 256) {
            int ii = idx / (TJ + 2), jj = idx - ii * (TJ + 2);
            int gi = min(max(i0 + ii - 1, 0), Hh - 1);
            int gj = min(max(j0 + jj - 1, 0), Ww - 1);
            int gp = gi * Ww + gj;
            lf[idx] = f[fb + (size_t)c * HW + gp];
#pragma unroll
            for (int n = 0; n < Nn; ++n)
                ln[n][idx] = nb[nbb + ((size_t)n * Cc + c) * HW + gp];
        }
        __syncthreads();
        if (valid) {
#pragma unroll
            for (int di = 0; di < 3; ++di)
#pragma unroll
                for (int dj = 0; dj < 3; ++dj) {
                    int li = (ti + di) * (TJ + 2) + (tj + dj);
                    bool t1 = lf[li] >= a1;
#pragma unroll
                    for (int n = 0; n < Nn; ++n)
                        cnt[n] += ((ln[n][li] >= a2[n]) == t1) ? 1 : 0;
                }
        }
        __syncthreads();
    }
    if (valid) {
#pragma unroll
        for (int n = 0; n < Nn; ++n)
            sim[((size_t)b * Nn + n) * HW + p] = (float)cnt[n];
    }
}

// ---------------------------------------------------------------------------
// 4) Softmax over N=4 per pixel
// ---------------------------------------------------------------------------
__global__ __launch_bounds__(256) void k_softmax(const float* __restrict__ sim,
                                                 float* __restrict__ wsm) {
    int idx = blockIdx.x * 256 + threadIdx.x;          // over B*HW
    if (idx >= Bb * HW) return;
    int b = idx / HW, p = idx - b * HW;
    float s[Nn], m = -3.4e38f;
#pragma unroll
    for (int n = 0; n < Nn; ++n) { s[n] = sim[((size_t)b * Nn + n) * HW + p]; m = fmaxf(m, s[n]); }
    float e[Nn], t = 0.f;
#pragma unroll
    for (int n = 0; n < Nn; ++n) { e[n] = __expf(s[n] - m); t += e[n]; }
    float inv = 1.0f / t;
#pragma unroll
    for (int n = 0; n < Nn; ++n) wsm[((size_t)b * Nn + n) * HW + p] = e[n] * inv;
}

// ---------------------------------------------------------------------------
// 5) w_fuse -> bf16
// ---------------------------------------------------------------------------
__global__ __launch_bounds__(256) void k_wcast(const float* __restrict__ w,
                                               __hip_bfloat16* __restrict__ wq) {
    int idx = blockIdx.x * 256 + threadIdx.x;
    if (idx < Cc * Cc) wq[idx] = __float2bfloat16(w[idx]);
}

// ---------------------------------------------------------------------------
// 6) fused = features + sum_n w[n]*nearby[n], cast to bf16, transpose to
//    pixel-major xq[P][C] via LDS (padded rows), 128-bit coalesced stores.
// ---------------------------------------------------------------------------
constexpr int ROWB = Cc + 8;                           // padded row (bf16 elems)

__global__ __launch_bounds__(256) void k_fuse(const float* __restrict__ f,
                                              const float* __restrict__ nb,
                                              const float* __restrict__ wsm,
                                              __hip_bfloat16* __restrict__ xq) {
    __shared__ __align__(16) __hip_bfloat16 buf[64 * ROWB];
    const int p0 = blockIdx.x * 64;                    // b taken from p0 (64 does not divide HW)
    const int b  = p0 / HW;
    const int hw0 = p0 - b * HW;
    const int pl = threadIdx.x & 63;
    const int c0 = threadIdx.x >> 6;                   // 0..3
    const int hw = hw0 + pl;

    float w[Nn];
#pragma unroll
    for (int n = 0; n < Nn; ++n) w[n] = wsm[((size_t)b * Nn + n) * HW + hw];

    const float* fp = f + ((size_t)b * Cc) * HW + hw;
    const float* np[Nn];
#pragma unroll
    for (int n = 0; n < Nn; ++n) np[n] = nb + (((size_t)b * Nn + n) * Cc) * HW + hw;

    for (int cc = c0; cc < Cc; cc += 4) {
        size_t off = (size_t)cc * HW;
        float v = fp[off];
#pragma unroll
        for (int n = 0; n < Nn; ++n) v += w[n] * np[n][off];
        buf[pl * ROWB + cc] = __float2bfloat16(v);
    }
    __syncthreads();
    const uint4* src = (const uint4*)buf;              // 8 bf16 per uint4
    uint4* dst = (uint4*)(xq + (size_t)p0 * Cc);
    for (int t = threadIdx.x; t < 64 * (Cc / 8); t += 256) {
        int row = t >> 5, qi = t & 31;                 // Cc/8 = 32 uint4/row
        dst[row * 32 + qi] = src[row * (ROWB / 8) + qi];
    }
}

// ---------------------------------------------------------------------------
// 7) 1x1 conv as bf16 WMMA GEMM, register-blocked over O:
//    each wave computes 4 O-tiles (64 out-channels) x 1 pixel tile (16 px),
//    so the activation B-fragment is loaded once per k-step and reused by
//    4 v_wmma_f32_16x16x32_bf16 -> activation traffic cut 4x (L2-resident).
//    Fragment layout per ISA §7.12.2: lane l<16 holds row l, K=k..k+7
//    (elems 0-7) and k+16..k+23 (elems 8-15); lanes 16-31 hold K=k+8.. and
//    k+24.. -> two contiguous b128 loads per fragment per lane.
// ---------------------------------------------------------------------------
constexpr int OB = 4;                                  // O-tiles per wave

__global__ __launch_bounds__(128) void k_gemm(const __hip_bfloat16* __restrict__ wq,
                                              const __hip_bfloat16* __restrict__ xq,
                                              const float* __restrict__ bias,
                                              float* __restrict__ out) {
    const int wave = threadIdx.x >> 5;                 // 0..3
    const int lane = threadIdx.x & 31;
    const int half = lane >> 4;
    const int l    = lane & 15;
    const int p0 = (blockIdx.x * 4 + wave) * 16;       // pixel tile base
    const int o0 = blockIdx.y * (OB * 16);             // 64 out-channels base

    union Frag { v16bf v; uint4 q[2]; };
    const __hip_bfloat16* xrow = xq + (size_t)(p0 + l) * Cc;
    const __hip_bfloat16* wrow[OB];
#pragma unroll
    for (int ot = 0; ot < OB; ++ot)
        wrow[ot] = wq + (size_t)(o0 + ot * 16 + l) * Cc;

    v8f acc[OB] = {};
#pragma unroll
    for (int kk = 0; kk < Cc; kk += 32) {
        Frag bfr;
        bfr.q[0] = *(const uint4*)(xrow + kk + 8 * half);
        bfr.q[1] = *(const uint4*)(xrow + kk + 16 + 8 * half);
        if (kk + 32 < Cc) __builtin_prefetch(xrow + kk + 32, 0, 0);
#pragma unroll
        for (int ot = 0; ot < OB; ++ot) {
            Frag a;
            a.q[0] = *(const uint4*)(wrow[ot] + kk + 8 * half);
            a.q[1] = *(const uint4*)(wrow[ot] + kk + 16 + 8 * half);
            acc[ot] = __builtin_amdgcn_wmma_f32_16x16x32_bf16(
                /*neg_a=*/false, a.v, /*neg_b=*/false, bfr.v,
                /*c_mod=*/(short)0, acc[ot], /*reuse_a=*/false, /*reuse_b=*/false);
        }
    }

    const int p  = p0 + l;                             // this lane's pixel col
    const int b  = p / HW;
    const int hw = p - b * HW;
#pragma unroll
    for (int ot = 0; ot < OB; ++ot) {
#pragma unroll
        for (int r = 0; r < 8; ++r) {
            int o = o0 + ot * 16 + r + 8 * half;       // C/D layout: M = r + 8*half
            out[((size_t)b * Cc + o) * HW + hw] = acc[ot][r] + bias[o];
        }
    }
}

// ---------------------------------------------------------------------------
extern "C" void kernel_launch(void* const* d_in, const int* in_sizes, int n_in,
                              void* d_out, int out_size, void* d_ws, size_t ws_size,
                              hipStream_t stream) {
    (void)in_sizes; (void)n_in; (void)out_size; (void)ws_size;
    const float* f   = (const float*)d_in[0];          // [B,C,H,W]
    const float* nb  = (const float*)d_in[1];          // [B,N,C,H,W]
    const float* wf  = (const float*)d_in[2];          // [C,C]
    const float* bf  = (const float*)d_in[3];          // [C]
    float* out = (float*)d_out;

    char* ws = (char*)d_ws;
    auto al = [](size_t x) { return (x + 255) & ~(size_t)255; };
    size_t off = 0;
    float* cs  = (float*)(ws + off); off = al(off + (size_t)5 * Bb * HW * 4);
    float* avg = (float*)(ws + off); off = al(off + (size_t)5 * Bb * HW * 4);
    float* sim = (float*)(ws + off); off = al(off + (size_t)Bb * Nn * HW * 4);
    float* wsm = (float*)(ws + off); off = al(off + (size_t)Bb * Nn * HW * 4);
    __hip_bfloat16* wq = (__hip_bfloat16*)(ws + off); off = al(off + (size_t)Cc * Cc * 2);
    __hip_bfloat16* xq = (__hip_bfloat16*)(ws + off); off = al(off + (size_t)PIX * Cc * 2);

    // 1) channel sums
    {
        int total = 5 * Bb * HW;
        k_chansum<<<(total + 255) / 256, 256, 0, stream>>>(f, nb, cs);
    }
    // 2) 3x3 box averages
    {
        int total = 5 * Bb * HW;
        k_avg<<<(total + 255) / 256, 256, 0, stream>>>(cs, avg);
    }
    // 3) census similarity
    {
        dim3 grid((Ww + TJ - 1) / TJ, (Hh + TI - 1) / TI, Bb);
        k_sim<<<grid, 256, 0, stream>>>(f, nb, avg, sim);
    }
    // 4) softmax over N
    {
        int total = Bb * HW;
        k_softmax<<<(total + 255) / 256, 256, 0, stream>>>(sim, wsm);
    }
    // 5) weights -> bf16
    k_wcast<<<(Cc * Cc + 255) / 256, 256, 0, stream>>>(wf, wq);
    // 6) fusion + bf16 transpose to [P][C]
    k_fuse<<<PIX / 64, 256, 0, stream>>>(f, nb, wsm, xq);
    // 7) WMMA GEMM -> output (each wave: 16 px x 64 out-channels)
    {
        dim3 grid(PIX / 64, Cc / (OB * 16));           // 950 x 4, 4 waves/block
        k_gemm<<<grid, 128, 0, stream>>>(wq, xq, bf, out);
    }
}
